// HyperConv2D_24266565222579
// MI455X (gfx1250) — matrix operand order repaired and checked by reading, not compiled, not run on hardware
//
#include <hip/hip_runtime.h>

typedef __attribute__((ext_vector_type(16))) __bf16          v16bf;
typedef __attribute__((ext_vector_type(16))) unsigned short  v16u;
typedef __attribute__((ext_vector_type(8)))  unsigned short  v8u;
typedef __attribute__((ext_vector_type(8)))  float           v8f;

#define B_   32
#define O_   64
#define I_   64
#define F_   16
#define H_   128
#define W_   128
#define KC_  18     // K = 9*64 = 576 = 18 chunks of 32
#define LDSC 72     // padded channel dimension in LDS (16B-aligned, conflict-free b128)

static __device__ __forceinline__ unsigned short f32_bf16(float f) {
  unsigned u = __float_as_uint(f);
  u += 0x7FFFu + ((u >> 16) & 1u);           // round-to-nearest-even
  return (unsigned short)(u >> 16);
}

// ---------------------------------------------------------------------------
// Kernel 1: softmax-mix the filter bank into per-(b,o,i) 3x3 weights, scattered
// directly into CDNA5 WMMA A-fragment layout (bf16), plus per-(b,o) biases.
// A-fragment layout (16-bit A 16x32): lane = ((K>>3)&1)*16 + M,
//                                     element = (K&7) | ((K>>4)<<3)
// K ordering: K = kl*64 + i  (kl = 3x3 tap index, i = input channel)
// ---------------------------------------------------------------------------
__global__ __launch_bounds__(64)
void hyper_wgen(const float* __restrict__ addr,     // (B,O,65,16)
                const float* __restrict__ w_bank,   // (16,3,3)
                const float* __restrict__ b_bank,   // (16)
                unsigned short* __restrict__ afrag, // [B][4 mtiles][18 kc][32 lane][16 elem]
                float* __restrict__ biasbuf) {      // [B*O]
  __shared__ float wb[F_ * 9];
  __shared__ float bb[F_];
  int bo = blockIdx.x;                 // 0..2047  = b*64 + o
  int b  = bo >> 6;
  int o  = bo & 63;
  int i  = threadIdx.x;                // input channel 0..63

  for (int t = threadIdx.x; t < F_ * 9; t += 64) wb[t] = w_bank[t];
  if (threadIdx.x < F_) bb[threadIdx.x] = b_bank[threadIdx.x];
  __syncthreads();

  const float* ap = addr + (((size_t)bo * 65) + 1 + i) * F_;
  float a[F_];
  float m = -1e30f;
#pragma unroll
  for (int f = 0; f < F_; ++f) { a[f] = ap[f]; m = fmaxf(m, a[f]); }
  float s = 0.f;
#pragma unroll
  for (int f = 0; f < F_; ++f) { a[f] = __expf(a[f] - m); s += a[f]; }
  float inv = 1.f / s;

  int mtile = o >> 4, mrow = o & 15;
#pragma unroll
  for (int kl = 0; kl < 9; ++kl) {
    float w = 0.f;
#pragma unroll
    for (int f = 0; f < F_; ++f) w += a[f] * wb[f * 9 + kl];
    w *= inv;
    int kc   = kl * 2 + (i >> 5);
    int k32  = i & 31;
    int e    = (k32 & 7) | ((k32 >> 4) << 3);
    int lane = ((k32 >> 3) & 1) * 16 + mrow;
    size_t pos = (((size_t)(b * 4 + mtile) * KC_) + kc) * 512 + lane * 16 + e;
    afrag[pos] = f32_bf16(w);
  }

  if (i == 0) {                              // bias for this (b,o)
    const float* bp = addr + ((size_t)bo * 65) * F_;
    float t[F_], mb = -1e30f;
    for (int f = 0; f < F_; ++f) { t[f] = bp[f]; mb = fmaxf(mb, t[f]); }
    float sb = 0.f, dot = 0.f;
    for (int f = 0; f < F_; ++f) { float e2 = __expf(t[f] - mb); sb += e2; dot += e2 * bb[f]; }
    biasbuf[bo] = dot / sb;
  }
}

// ---------------------------------------------------------------------------
// Kernel 2: implicit-GEMM conv, one block per (batch, output row).
//  - A fragments for this batch (72KB, bf16, fragment-ready) are staged into
//    LDS with global_load_async_to_lds_b128 (ASYNCcnt), overlapped with the
//    f32->bf16 staging of the 3-row x halo (channel-last xs[3][132][72]).
//  - 8 waves: wave w -> mtile (w&3), n-half (w>>2); 4 f32 accumulators each,
//    18 k-chunks of back-to-back v_wmma_f32_16x16x32_bf16.
// ---------------------------------------------------------------------------
__global__ __launch_bounds__(256)
void hyper_conv(const float* __restrict__ x,            // (B,I,H,W)
                const unsigned short* __restrict__ afrag,
                const float* __restrict__ biasbuf,
                float* __restrict__ out) {              // (B,O,H,W)
  __shared__ unsigned short xs[3][132][LDSC];                   // 57,024 B
  __shared__ __align__(32) unsigned short as_a[4 * KC_ * 512];  // 73,728 B

  int blk = blockIdx.x;
  int b   = blk >> 7;          // /128
  int ph  = blk & 127;         // output row
  int tid = threadIdx.x;

  // ---- async DMA: afrag[b] (73,728 B) -> LDS, 16B per lane, 18 issues ----
  {
    const unsigned short* g = afrag + (size_t)(b * 4) * KC_ * 512;
#pragma unroll 1
    for (int it = 0; it < 18; ++it) {
      int off = (it * 256 + tid) * 8;                 // ushort index (16B/lane)
      unsigned lds = (unsigned)(size_t)(&as_a[off]);  // LDS byte offset
      const unsigned short* ga = g + off;
      asm volatile("global_load_async_to_lds_b128 %0, %1, off"
                   :: "v"(lds), "v"(ga) : "memory");
    }
  }

  // ---- zero xs (halo cols + out-of-range rows) ----
  for (int t = tid; t < 3 * 132 * LDSC; t += 256)
    ((unsigned short*)xs)[t] = 0;
  __syncthreads();

  // ---- stage x[b, :, ph-1..ph+1, :] -> bf16, channel-last ----
  {
    int colt = tid & 127;
    int i2   = tid >> 7;       // 0..1
    for (int r = 0; r < 3; ++r) {
      int y = ph + r - 1;
      if ((unsigned)y < (unsigned)H_) {
        for (int ib = 0; ib < I_; ib += 2) {
          int ch  = ib + i2;
          float v = x[(((size_t)b * I_ + ch) * H_ + y) * W_ + colt];
          xs[r][colt + 1][ch] = f32_bf16(v);
        }
      }
    }
  }
  asm volatile("s_wait_asynccnt 0x0" ::: "memory");
  __syncthreads();

  int wave  = tid >> 5;
  int lane  = tid & 31;
  int mtile = wave & 3;        // group of 16 output channels
  int ngrp  = wave >> 2;       // half of the 128 output columns
  int ln15  = lane & 15;
  int lhi   = lane >> 4;

  v8f acc0 = {}, acc1 = {}, acc2 = {}, acc3 = {};
  const unsigned short* la0 = as_a + (size_t)mtile * KC_ * 512 + lane * 16;

#pragma unroll 2
  for (int kc = 0; kc < KC_; ++kc) {
    v16u  au = *(const v16u*)(la0 + kc * 512);        // 32B, 32B-aligned LDS
    v16bf av = __builtin_bit_cast(v16bf, au);

    int kl = kc >> 1;
    int dk = kl / 3, dl = kl - dk * 3;                // 3x3 tap offsets
    int i0 = (kc & 1) * 32 + lhi * 8;                 // channels i0..i0+7 and i0+16..i0+23
    int col = ngrp * 64 + ln15 + dl;                  // halo-shifted LDS column
    const unsigned short* bp = &xs[dk][col][i0];

    // batch all 8 B-fragment b128 loads, then 4 back-to-back WMMAs
    v8u p0 = *(const v8u*)(bp);                 v8u q0 = *(const v8u*)(bp + 16);
    v8u p1 = *(const v8u*)(bp + 16 * LDSC);     v8u q1 = *(const v8u*)(bp + 16 * LDSC + 16);
    v8u p2 = *(const v8u*)(bp + 32 * LDSC);     v8u q2 = *(const v8u*)(bp + 32 * LDSC + 16);
    v8u p3 = *(const v8u*)(bp + 48 * LDSC);     v8u q3 = *(const v8u*)(bp + 48 * LDSC + 16);
    v16u u0, u1, u2, u3;
#pragma unroll
    for (int e = 0; e < 8; ++e) {
      u0[e] = p0[e]; u0[e + 8] = q0[e];
      u1[e] = p1[e]; u1[e + 8] = q1[e];
      u2[e] = p2[e]; u2[e + 8] = q2[e];
      u3[e] = p3[e]; u3[e + 8] = q3[e];
    }
    acc0 = __builtin_amdgcn_wmma_f32_16x16x32_bf16(
        false, av, false, __builtin_bit_cast(v16bf, u0), (short)0, acc0, false, false);
    acc1 = __builtin_amdgcn_wmma_f32_16x16x32_bf16(
        false, av, false, __builtin_bit_cast(v16bf, u1), (short)0, acc1, false, false);
    acc2 = __builtin_amdgcn_wmma_f32_16x16x32_bf16(
        false, av, false, __builtin_bit_cast(v16bf, u2), (short)0, acc2, false, false);
    acc3 = __builtin_amdgcn_wmma_f32_16x16x32_bf16(
        false, av, false, __builtin_bit_cast(v16bf, u3), (short)0, acc3, false, false);
  }

  // ---- epilogue: bias add + store (lanes 0-15 write 64B-contiguous runs) ----
  int obase = mtile * 16 + lhi * 8;                   // 8 consecutive out-channels
  const float* bias = biasbuf + b * 64 + obase;
  float bv[8];
#pragma unroll
  for (int r = 0; r < 8; ++r) bv[r] = bias[r];

  float* ob = out + (((size_t)b * O_ + obase) * H_ + ph) * W_ + ngrp * 64 + ln15;
#pragma unroll
  for (int r = 0; r < 8; ++r) {
    float* orow = ob + (size_t)r * H_ * W_;
    orow[0]  = acc0[r] + bv[r];
    orow[16] = acc1[r] + bv[r];
    orow[32] = acc2[r] + bv[r];
    orow[48] = acc3[r] + bv[r];
  }
}

// ---------------------------------------------------------------------------
extern "C" void kernel_launch(void* const* d_in, const int* in_sizes, int n_in,
                              void* d_out, int out_size, void* d_ws, size_t ws_size,
                              hipStream_t stream) {
  (void)in_sizes; (void)n_in; (void)out_size; (void)ws_size;
  const float* addresses = (const float*)d_in[0];   // (32,64,65,16)
  const float* x         = (const float*)d_in[1];   // (32,64,128,128)
  const float* w_bank    = (const float*)d_in[2];   // (16,3,3)
  const float* b_bank    = (const float*)d_in[3];   // (16)
  // d_in[4]=s(1), d_in[5]=p(1): compile-time constants here.

  unsigned short* afrag = (unsigned short*)d_ws;    // 128*18*512*2 = 2.25 MB
  float* biasbuf = (float*)((char*)d_ws +
                   (size_t)(B_ * 4) * KC_ * 512 * sizeof(unsigned short));
  float* out = (float*)d_out;

  hipLaunchKernelGGL(hyper_wgen, dim3(B_ * O_), dim3(64), 0, stream,
                     addresses, w_bank, b_bank, afrag, biasbuf);
  hipLaunchKernelGGL(hyper_conv, dim3(B_ * H_), dim3(256), 0, stream,
                     x, afrag, biasbuf, out);
}
